// MultiHeadAttention_39118562132226
// MI455X (gfx1250) — matrix operand order repaired and checked by reading, not compiled, not run on hardware
//
#include <hip/hip_runtime.h>
#include <hip/hip_bf16.h>
#include <math.h>

// ---------------- types ----------------
typedef __bf16 bf16_t;
typedef __attribute__((ext_vector_type(16))) __bf16 v16bf;
typedef __attribute__((ext_vector_type(8)))  __bf16 v8bf;
typedef __attribute__((ext_vector_type(8)))  float  v8f;

union Frag { v16bf v; v8bf h[2]; bf16_t e[16]; };

__device__ __forceinline__ bf16_t to_bf16(float f) {
  union { float f; unsigned u; } in; in.f = f;
  unsigned r = in.u + 0x7FFFu + ((in.u >> 16) & 1u);   // RNE
  union { unsigned short s; bf16_t b; } out;
  out.s = (unsigned short)(r >> 16);
  return out.b;
}

// Problem constants
#define BB   16
#define TT   2048
#define HH   6
#define DD   64
#define CC   384
#define MM   (BB*TT)          // 32768
#define NQKV (3*HH*DD)        // 1152

// 1/sqrt(64) * log2(e): scores come out of WMMA already in exp2 domain
#define QSCALE 0.18033688011112043f

// ---------------- conversion kernels ----------------
__global__ void conv_x_kernel(const float* __restrict__ x, bf16_t* __restrict__ xb, int n) {
  int i = blockIdx.x * 256 + threadIdx.x;
  if (i < n) xb[i] = to_bf16(x[i]);
}

// WT:  [1152][384]  row n = output col (q:0-383 | k:384-767 | v:768-1151), col = k
// WoT: [384][384]   row n, col k  (transpose of Wo)
__global__ void conv_w_kernel(const float* __restrict__ wq, const float* __restrict__ wk,
                              const float* __restrict__ wv, const float* __restrict__ wo,
                              bf16_t* __restrict__ wt, bf16_t* __restrict__ wot) {
  int i = blockIdx.x * 256 + threadIdx.x;
  const int nwt = NQKV * CC;            // 442368
  const int nwo = CC * CC;              // 147456
  if (i < nwt) {
    int n = i / CC, k = i % CC;
    int p = n / CC;                     // 0=q,1=k,2=v   (n in [0,1152))
    int within = n % CC;
    int h = within / DD, d = within % DD;
    const float* src = (p == 0) ? wq : (p == 1) ? wk : wv;  // [H][C][D]
    wt[i] = to_bf16(src[((size_t)h * CC + k) * DD + d]);
  } else if (i < nwt + nwo) {
    int j = i - nwt;
    int n = j / CC, k = j % CC;
    wot[j] = to_bf16(wo[(size_t)k * CC + n]);
  }
}

// ---------------- QKV projection GEMM ----------------
// M=32768, K=384, N=1152.  8 waves/block, each wave = 16x64 tile.
// Column group gy (64 cols) == one (proj, head) pair.  V stored transposed [b,h,d,t].
// Q is pre-scaled by 0.125*log2(e) so attention scores land in the exp2 domain.
__global__ __launch_bounds__(256) void qkv_gemm_kernel(const bf16_t* __restrict__ xb,
    const bf16_t* __restrict__ wt, bf16_t* __restrict__ qws,
    bf16_t* __restrict__ kws, bf16_t* __restrict__ vtws) {
  const int lane = threadIdx.x & 31;
  const int wave = threadIdx.x >> 5;
  const int m0   = (blockIdx.x * 8 + wave) * 16;
  const int gy   = blockIdx.y;            // 0..17
  const int nbase = gy * 64;
  const int lo16 = lane & 15, hi = lane >> 4;
  const int koff = hi * 16;

  v8f acc[4] = {};
  const bf16_t* arow = xb + (size_t)(m0 + lo16) * CC;
  #pragma unroll
  for (int kb = 0; kb < CC; kb += 32) {
    Frag a;
    a.h[0] = *(const v8bf*)(arow + kb + koff);
    a.h[1] = *(const v8bf*)(arow + kb + 8 + koff);
    #pragma unroll
    for (int nt = 0; nt < 4; ++nt) {
      const int n = nbase + nt * 16 + lo16;
      v16bf b = *(const v16bf*)(wt + (size_t)n * CC + kb + hi * 16);
      acc[nt] = __builtin_amdgcn_wmma_f32_16x16x32_bf16(false, a.v, false, b,
                                                        (short)0, acc[nt], false, false);
    }
  }
  const int proj = gy / HH;               // 0=q,1=k,2=v
  const int h_   = gy % HH;
  const float sc = (proj == 0) ? QSCALE : 1.0f;   // wave-uniform
  #pragma unroll
  for (int nt = 0; nt < 4; ++nt) {
    #pragma unroll
    for (int i = 0; i < 8; ++i) {
      int m  = m0 + i + 8 * hi;
      int b_ = m >> 11, t = m & (TT - 1);
      int d  = nt * 16 + lo16;
      size_t bh = (size_t)b_ * HH + h_;
      bf16_t val = to_bf16(acc[nt][i] * sc);
      if (proj == 0)      qws[(bh * TT + t) * DD + d] = val;
      else if (proj == 1) kws[(bh * TT + t) * DD + d] = val;
      else                vtws[(bh * DD + d) * TT + t] = val;
    }
  }
}

// ---------------- fused causal flash attention ----------------
// 8 waves/block; wave owns 16 query rows; block = 128 rows of one (b,h).
// Row-sum of P computed on the XDL pipe via P @ Ones (all 16 lanes of a half
// end up holding their row's normalizer) -- no shuffle-add chain.
__global__ __launch_bounds__(256) void flash_attn_kernel(const bf16_t* __restrict__ qws,
    const bf16_t* __restrict__ kws, const bf16_t* __restrict__ vtws,
    bf16_t* __restrict__ ows) {
  __shared__ bf16_t pbuf[8][16][32];      // 8 KB, per-wave P staging (C-layout -> A-frag)
  const int lane = threadIdx.x & 31;
  const int wave = threadIdx.x >> 5;
  const int bh = blockIdx.y;              // 0..95
  const int b_ = bh / HH, h_ = bh % HH;
  const int qr0 = blockIdx.x * 128 + wave * 16;
  const int lo16 = lane & 15, hi = lane >> 4;
  const int koff = hi * 16;

  const bf16_t* qp = qws  + (size_t)bh * TT * DD;
  const bf16_t* kp = kws  + (size_t)bh * TT * DD;
  const bf16_t* vp = vtws + (size_t)bh * DD * TT;

  Frag qf[2];
  {
    const bf16_t* qrow = qp + (size_t)(qr0 + lo16) * DD;
    #pragma unroll
    for (int c = 0; c < 2; ++c) {
      qf[c].h[0] = *(const v8bf*)(qrow + c * 32 + koff);
      qf[c].h[1] = *(const v8bf*)(qrow + c * 32 + 8 + koff);
    }
  }

  Frag ones;                              // 32x16 all-ones B fragment
  #pragma unroll
  for (int e = 0; e < 16; ++e) ones.e[e] = to_bf16(1.0f);

  v8f accO[4] = {};
  v8f lacc   = {};                        // row-sum accumulator (P @ Ones)
  float mrow[8];
  #pragma unroll
  for (int i = 0; i < 8; ++i) mrow[i] = -3.0e38f;

  const int send = qr0 + 16;              // keys needed: <= qr0+15 (wave-uniform bound)
  for (int sblk = 0; sblk < send; sblk += 32) {
    // ---- prefetch next key block while this one computes (wave-uniform guard) ----
    if (sblk + 32 < send) {
      __builtin_prefetch(kp + (size_t)(sblk + 32 + lane) * DD, 0, 1);
      __builtin_prefetch(vp + (size_t)(lane * 2) * TT + sblk + 32, 0, 1);
    }
    // ---- scores: two 16x16 f32 tiles over 32 keys, K-dim = d = 64 ----
    v8f s0 = {}, s1 = {};
    #pragma unroll
    for (int c = 0; c < 2; ++c) {
      v16bf k0 = *(const v16bf*)(kp + (size_t)(sblk + lo16) * DD + c * 32 + hi * 16);
      v16bf k1 = *(const v16bf*)(kp + (size_t)(sblk + 16 + lo16) * DD + c * 32 + hi * 16);
      s0 = __builtin_amdgcn_wmma_f32_16x16x32_bf16(false, qf[c].v, false, k0, (short)0, s0, false, false);
      s1 = __builtin_amdgcn_wmma_f32_16x16x32_bf16(false, qf[c].v, false, k1, (short)0, s1, false, false);
    }
    const bool need_mask = (sblk + 31) > qr0;   // wave-uniform
    // ---- online softmax (row = i + 8*hi spans the 16 lanes of a half) ----
    #pragma unroll
    for (int i = 0; i < 8; ++i) {
      int qrow = qr0 + i + 8 * hi;
      float v0 = s0[i];                    // already scaled into exp2 domain
      float v1 = s1[i];
      if (need_mask) {
        if (sblk + lo16 > qrow)      v0 = -3.0e38f;
        if (sblk + 16 + lo16 > qrow) v1 = -3.0e38f;
      }
      float rmax = fmaxf(v0, v1);
      #pragma unroll
      for (int off = 1; off < 16; off <<= 1)
        rmax = fmaxf(rmax, __shfl_xor(rmax, off, 32));
      float mnew  = fmaxf(mrow[i], rmax);
      float alpha = exp2f(mrow[i] - mnew);
      float p0 = exp2f(v0 - mnew);
      float p1 = exp2f(v1 - mnew);
      mrow[i] = mnew;
      lacc[i] *= alpha;
      accO[0][i] *= alpha; accO[1][i] *= alpha;
      accO[2][i] *= alpha; accO[3][i] *= alpha;
      pbuf[wave][i + 8 * hi][lo16]      = to_bf16(p0);
      pbuf[wave][i + 8 * hi][16 + lo16] = to_bf16(p1);
    }
    // ---- P (C-layout) -> A-fragment via per-wave LDS (same-wave, in-order) ----
    Frag pa;
    pa.h[0] = *(const v8bf*)&pbuf[wave][lo16][koff];
    pa.h[1] = *(const v8bf*)&pbuf[wave][lo16][8 + koff];
    // ---- O += P[16x32] @ V[32x64]; rowsum += P @ Ones (on the matrix pipe) ----
    lacc = __builtin_amdgcn_wmma_f32_16x16x32_bf16(false, pa.v, false, ones.v,
                                                   (short)0, lacc, false, false);
    #pragma unroll
    for (int nt = 0; nt < 4; ++nt) {
      v16bf vb = *(const v16bf*)(vp + (size_t)(nt * 16 + lo16) * TT + sblk + hi * 16);
      accO[nt] = __builtin_amdgcn_wmma_f32_16x16x32_bf16(false, pa.v, false, vb,
                                                         (short)0, accO[nt], false, false);
    }
  }

  // ---- epilogue: normalize, store concat-head O as bf16 [32768 x 384] ----
  // lacc's C-tile has the row sum replicated across all 16 columns -> every
  // lane already holds its row's normalizer.
  #pragma unroll
  for (int nt = 0; nt < 4; ++nt) {
    #pragma unroll
    for (int i = 0; i < 8; ++i) {
      int t = qr0 + i + 8 * hi;
      float o = accO[nt][i] / lacc[i];
      ows[((size_t)b_ * TT + t) * CC + h_ * DD + nt * 16 + lo16] = to_bf16(o);
    }
  }
}

// ---------------- output projection GEMM (+bias, f32 out) ----------------
__global__ __launch_bounds__(256) void out_gemm_kernel(const bf16_t* __restrict__ ob,
    const bf16_t* __restrict__ wot, const float* __restrict__ bo,
    float* __restrict__ out) {
  const int lane = threadIdx.x & 31;
  const int wave = threadIdx.x >> 5;
  const int m0 = (blockIdx.x * 8 + wave) * 16;
  const int nbase = blockIdx.y * 64;
  const int lo16 = lane & 15, hi = lane >> 4, koff = hi * 16;

  v8f acc[4] = {};
  const bf16_t* arow = ob + (size_t)(m0 + lo16) * CC;
  #pragma unroll
  for (int kb = 0; kb < CC; kb += 32) {
    Frag a;
    a.h[0] = *(const v8bf*)(arow + kb + koff);
    a.h[1] = *(const v8bf*)(arow + kb + 8 + koff);
    #pragma unroll
    for (int nt = 0; nt < 4; ++nt) {
      v16bf b = *(const v16bf*)(wot + (size_t)(nbase + nt * 16 + lo16) * CC + kb + hi * 16);
      acc[nt] = __builtin_amdgcn_wmma_f32_16x16x32_bf16(false, a.v, false, b,
                                                        (short)0, acc[nt], false, false);
    }
  }
  #pragma unroll
  for (int nt = 0; nt < 4; ++nt) {
    float bias = bo[nbase + nt * 16 + lo16];
    #pragma unroll
    for (int i = 0; i < 8; ++i)
      out[(size_t)(m0 + i + 8 * hi) * CC + nbase + nt * 16 + lo16] = acc[nt][i] + bias;
  }
}

// ---------------- host launch ----------------
extern "C" void kernel_launch(void* const* d_in, const int* in_sizes, int n_in,
                              void* d_out, int out_size, void* d_ws, size_t ws_size,
                              hipStream_t stream) {
  (void)in_sizes; (void)n_in; (void)out_size; (void)ws_size;
  const float* x  = (const float*)d_in[0];
  const float* Wq = (const float*)d_in[1];
  const float* Wk = (const float*)d_in[2];
  const float* Wv = (const float*)d_in[3];
  const float* Wo = (const float*)d_in[4];
  const float* bo = (const float*)d_in[5];
  float* out = (float*)d_out;

  char* ws = (char*)d_ws;
  size_t off = 0;
  auto carve = [&](size_t bytes) -> void* {
    void* p = (void*)(ws + off);
    off += (bytes + 255) & ~(size_t)255;
    return p;
  };
  bf16_t* xb  = (bf16_t*)carve((size_t)MM * CC * 2);          // 25.2 MB
  bf16_t* wt  = (bf16_t*)carve((size_t)NQKV * CC * 2);        // 0.9 MB
  bf16_t* wot = (bf16_t*)carve((size_t)CC * CC * 2);          // 0.3 MB
  bf16_t* qws = (bf16_t*)carve((size_t)BB * HH * TT * DD * 2);// 25.2 MB
  bf16_t* kws = (bf16_t*)carve((size_t)BB * HH * TT * DD * 2);// 25.2 MB
  bf16_t* vt  = (bf16_t*)carve((size_t)BB * HH * DD * TT * 2);// 25.2 MB
  bf16_t* ob  = (bf16_t*)carve((size_t)MM * CC * 2);          // 25.2 MB

  const int nx = MM * CC;
  conv_x_kernel<<<(nx + 255) / 256, 256, 0, stream>>>(x, xb, nx);
  const int nw = NQKV * CC + CC * CC;
  conv_w_kernel<<<(nw + 255) / 256, 256, 0, stream>>>(Wq, Wk, Wv, Wo, wt, wot);

  // QKV: 2048 M-tiles / 8 waves = 256 blocks; 18 col groups
  qkv_gemm_kernel<<<dim3(256, 18), 256, 0, stream>>>(xb, wt, qws, kws, vt);
  // Flash attention: 16 q-blocks of 128 rows; 96 (b,h) pairs
  flash_attn_kernel<<<dim3(16, 96), 256, 0, stream>>>(qws, kws, vt, ob);
  // Output projection: 256 M-blocks; 6 col groups
  out_gemm_kernel<<<dim3(256, 6), 256, 0, stream>>>(ob, wot, bo, out);
}
